// ObjectModel_63608465653936
// MI455X (gfx1250) — matrix operand order repaired and checked by reading, not compiled, not run on hardware
//
#include <hip/hip_runtime.h>
#include <math.h>

// ---------------- problem constants ----------------
constexpr int Bn  = 16;
constexpr int Sn  = 512;
constexpr int Dn  = 512;
constexpr int Hn  = 8;
constexpr int DHn = 64;     // Dn / Hn
constexpr int FFn = 2048;
constexpr int Kc  = 64;     // #centres
constexpr int Ln  = 4;      // layers
constexpr int Ntok = Bn * Sn;           // 8192 token rows
constexpr float EPSc = 1e-5f;
constexpr float SCALEc = 0.125f;        // 1/sqrt(64)

// ---------------- CDNA5 vector types ----------------
typedef __attribute__((ext_vector_type(16))) __bf16 v16bf;
typedef __attribute__((ext_vector_type(8)))  __bf16 v8bf;
typedef __attribute__((ext_vector_type(8)))  float  v8f;

// Load one lane's A-operand (16xK=32 bf16 tile) with the CDNA5 half-lane K split:
// lanes 0-15 hold K = {0..7, 16..23}, lanes 16-31 hold K = {8..15, 24..31}.
static __device__ inline v16bf load_a16(const __bf16* __restrict__ row, int k0, int hi) {
    const v8bf lo  = *(const v8bf*)(row + k0 + hi * 8);
    const v8bf hi8 = *(const v8bf*)(row + k0 + 16 + hi * 8);
    v16bf a;
#pragma unroll
    for (int i = 0; i < 8; ++i) { a[i] = lo[i]; a[i + 8] = hi8[i]; }
    return a;
}

static __device__ inline v8f wmma_bf16(v16bf a, v16bf b, v8f c) {
    return __builtin_amdgcn_wmma_f32_16x16x32_bf16(false, a, false, b, (short)0, c,
                                                   false, false);
}

// ---------------- small precompute kernels ----------------
__global__ void k_centre_sums(const float* __restrict__ centre, float* __restrict__ sc) {
    int k = threadIdx.x;                 // 64 threads
    const float4* r = (const float4*)(centre + (size_t)k * Dn);
    float s = 0.f;
#pragma unroll 4
    for (int j = 0; j < Dn / 4; ++j) { float4 v = r[j]; s += v.x + v.y + v.z + v.w; }
    sc[k] = s;
}

__global__ void k_labels(const float* __restrict__ x, const float* __restrict__ sc,
                         int* __restrict__ lab) {
    int i = blockIdx.x * blockDim.x + threadIdx.x;
    if (i >= Ntok) return;
    const float4* r = (const float4*)(x + (size_t)i * Dn);
    float s = 0.f;
#pragma unroll 4
    for (int j = 0; j < Dn / 4; ++j) { float4 v = r[j]; s += v.x + v.y + v.z + v.w; }
    float best = __builtin_inff(); int bi = 0;
    for (int k = 0; k < Kc; ++k) {
        float d = fabsf(s - sc[k]);
        if (d < best) { best = d; bi = k; }
    }
    lab[i] = bi;
}

__global__ void k_init_x(const float* __restrict__ in, float* __restrict__ xf,
                         __bf16* __restrict__ xb, int n) {
    int i = blockIdx.x * blockDim.x + threadIdx.x;
    if (i >= n) return;
    float v = in[i];
    xf[i] = v;
    xb[i] = (__bf16)v;
}

// W[M][K] f32  ->  Wt[K][M] bf16  (so WMMA B-operand rows are contiguous)
__global__ void k_wt(const float* __restrict__ W, __bf16* __restrict__ Wt,
                     int Mdim, int Kdim) {
    int i = blockIdx.x * blockDim.x + threadIdx.x;
    if (i >= Mdim * Kdim) return;
    int m = i / Kdim, k = i - m * Kdim;
    Wt[(size_t)k * Mdim + m] = (__bf16)W[i];
}

// ---------------- WMMA GEMM: out[n,m] = sum_k A[n,k]*Bt[k,m] + bias[m] ----------------
// One wave computes a 32x64 output tile: 2 M-subtiles x 4 N-subtiles = 8 WMMA
// accumulators fed by 2 A-loads + 4 B-loads per K-step (~21 flop/byte).
enum { ST_F32 = 0, ST_BF16 = 1, ST_KT = 2 };

template <int MODE, bool RELU>
__global__ __launch_bounds__(256) void k_gemm(const __bf16* __restrict__ A,
                                              const __bf16* __restrict__ Bt,
                                              const float* __restrict__ bias,
                                              void* __restrict__ outp,
                                              int N, int M, int Kd) {
    const int lane = threadIdx.x & 31;
    const int wid  = threadIdx.x >> 5;
    const long wave = (long)blockIdx.x * 8 + wid;
    const int tilesM = M >> 6;                    // 64-wide N tiles
    const long tileCount = (long)(N >> 5) * tilesM;
    if (wave >= tileCount) return;
    const int tn = (int)(wave / tilesM);
    const int tm = (int)(wave - (long)tn * tilesM);
    const int n0 = tn << 5, m0 = tm << 6;
    const int hi = lane >> 4, lm = lane & 15;

    const __bf16* arow0 = A + (size_t)(n0 + lm) * Kd;
    const __bf16* arow1 = A + (size_t)(n0 + 16 + lm) * Kd;
    const __bf16* brow  = Bt + m0 + lm;   // column n = m0 + lm

    v8f acc[8];
#pragma unroll
    for (int i = 0; i < 8; ++i) acc[i] = v8f{0.f, 0.f, 0.f, 0.f, 0.f, 0.f, 0.f, 0.f};

    for (int k0 = 0; k0 < Kd; k0 += 32) {
        if (k0 + 64 < Kd) {   // cover L2 latency for the K-step after next
            __builtin_prefetch(arow0 + k0 + 64, 0, 1);
            __builtin_prefetch(arow1 + k0 + 64, 0, 1);
            __builtin_prefetch(brow + (size_t)(k0 + 64) * M, 0, 1);
        }
        v16bf a0 = load_a16(arow0, k0, hi);
        v16bf a1 = load_a16(arow1, k0, hi);
        const __bf16* bk = brow + (size_t)k0 * M;
#pragma unroll
        for (int nt = 0; nt < 4; ++nt) {
            v16bf bv;
#pragma unroll
            for (int s = 0; s < 16; ++s)
                bv[s] = bk[(size_t)(8 * hi + (s & 7) + 16 * (s >> 3)) * M + nt * 16];
            acc[nt]     = wmma_bf16(a0, bv, acc[nt]);
            acc[4 + nt] = wmma_bf16(a1, bv, acc[4 + nt]);
        }
    }

#pragma unroll
    for (int nt = 0; nt < 4; ++nt) {
        const int col = m0 + nt * 16 + lm;
        const float bz = bias[col];
#pragma unroll
        for (int half = 0; half < 2; ++half) {
#pragma unroll
            for (int i = 0; i < 8; ++i) {
                float v = acc[half * 4 + nt][i] + bz;
                if (RELU) v = fmaxf(v, 0.f);
                const int r = n0 + half * 16 + i + 8 * hi;   // global token row
                if (MODE == ST_F32) {
                    ((float*)outp)[(size_t)r * M + col] = v;
                } else if (MODE == ST_BF16) {
                    ((__bf16*)outp)[(size_t)r * M + col] = (__bf16)v;
                } else {                                      // K transposed: [B][D][S]
                    const int bIdx = r >> 9;                  // r / Sn
                    const int s    = r & (Sn - 1);
                    ((__bf16*)outp)[(size_t)bIdx * Dn * Sn + (size_t)col * Sn + s] =
                        (__bf16)v;
                }
            }
        }
    }
}

// ---------------- fused masked flash attention ----------------
// grid: Bn*Hn*(Sn/16) blocks of one wave32 each.
__global__ __launch_bounds__(32) void k_attn(const __bf16* __restrict__ qb,
                                             const __bf16* __restrict__ ktb,
                                             const __bf16* __restrict__ vb,
                                             const int* __restrict__ lab,
                                             __bf16* __restrict__ ob) {
    const int qt = blockIdx.x & 31;          // Sn/16 = 32 query tiles
    const int h  = (blockIdx.x >> 5) & (Hn - 1);
    const int b  = blockIdx.x >> 8;
    const int lane = threadIdx.x;
    const int hi = lane >> 4, lm = lane & 15;
    const int q0 = qt << 4;

    __shared__ __attribute__((aligned(64))) float  sd[16][33];   // raw scores (q, key)
    __shared__ __attribute__((aligned(64))) __bf16 pb[16][32];   // probs, bf16 A layout
    __shared__ float al_s[16], l_s[16];
    __shared__ int   kl[32];

    // per-row online-softmax state (owned by lanes 0..15; row == lane)
    float m_r = -__builtin_inff();
    float l_r = 0.f;
    int   myql = 0;
    if (lane < 16) myql = lab[b * Sn + q0 + lane];

    // preload this wave's Q tile (16 q rows x DH=64) as two K-step A operands
    const __bf16* qrow = qb + ((size_t)(b * Sn + q0 + lm)) * Dn + h * DHn;
    const v16bf qa0 = load_a16(qrow, 0, hi);
    const v16bf qa1 = load_a16(qrow, 32, hi);

    const __bf16* ktbase = ktb + ((size_t)(b * Dn + h * DHn)) * Sn;
    const __bf16* vbase  = vb + ((size_t)b * Sn) * Dn + h * DHn;

    v8f acc[4];
#pragma unroll
    for (int t = 0; t < 4; ++t) acc[t] = v8f{0.f, 0.f, 0.f, 0.f, 0.f, 0.f, 0.f, 0.f};

    for (int key0 = 0; key0 < Sn; key0 += 32) {
        kl[lane] = lab[b * Sn + key0 + lane];

        // --- scores: two 16-key tiles, K-loop over DH=64 in two steps of 32 ---
#pragma unroll
        for (int t = 0; t < 2; ++t) {
            v8f s = {0.f, 0.f, 0.f, 0.f, 0.f, 0.f, 0.f, 0.f};
            v16bf b0, b1;
#pragma unroll
            for (int e = 0; e < 16; ++e) {
                const int kk = 8 * hi + (e & 7) + 16 * (e >> 3);
                b0[e] = ktbase[(size_t)kk * Sn + key0 + t * 16 + lm];
                b1[e] = ktbase[(size_t)(32 + kk) * Sn + key0 + t * 16 + lm];
            }
            s = wmma_bf16(qa0, b0, s);

            s = wmma_bf16(qa1, b1, s);
#pragma unroll
            for (int i = 0; i < 8; ++i) sd[i + 8 * hi][lm + 16 * t] = s[i];
        }
        __syncthreads();

        // --- masked online softmax: lanes 0..15 each own one query row ---
        if (lane < 16) {
            float sv[32];
            float cm = -__builtin_inff();
#pragma unroll
            for (int j = 0; j < 32; ++j) {
                float v = sd[lane][j] * SCALEc;
                if (kl[j] != myql) v = -__builtin_inff();
                sv[j] = v;
                cm = fmaxf(cm, v);
            }
            float newm  = fmaxf(m_r, cm);
            float alpha = (newm == -__builtin_inff()) ? 1.f : __expf(m_r - newm);
            float psum  = 0.f;
#pragma unroll
            for (int j = 0; j < 32; ++j) {
                float p = (sv[j] == -__builtin_inff()) ? 0.f : __expf(sv[j] - newm);
                pb[lane][j] = (__bf16)p;
                psum += p;
            }
            l_r = l_r * alpha + psum;
            m_r = newm;
            al_s[lane] = alpha;
            l_s[lane]  = l_r;
        }
        __syncthreads();

        // --- rescale accumulators by per-row alpha ---
        float al[8];
#pragma unroll
        for (int i = 0; i < 8; ++i) al[i] = al_s[i + 8 * hi];
#pragma unroll
        for (int t = 0; t < 4; ++t)
#pragma unroll
            for (int i = 0; i < 8; ++i) acc[t][i] *= al[i];

        // --- P (16x32) @ V (32x64): 4 WMMAs over the 16-wide N tiles ---
        v16bf pa = load_a16(&pb[lm][0], 0, hi);
#pragma unroll
        for (int t = 0; t < 4; ++t) {
            v16bf vv;
#pragma unroll
            for (int e = 0; e < 16; ++e)
                vv[e] = vbase[(size_t)(key0 + 8 * hi + (e & 7) + 16 * (e >> 3)) * Dn + t * 16 + lm];
            acc[t] = wmma_bf16(pa, vv, acc[t]);
        }
    }

    // --- normalize by the softmax denominator and store bf16 output ---
    float inv[8];
#pragma unroll
    for (int i = 0; i < 8; ++i) inv[i] = 1.f / l_s[i + 8 * hi];
#pragma unroll
    for (int t = 0; t < 4; ++t)
#pragma unroll
        for (int i = 0; i < 8; ++i) {
            int q = q0 + i + 8 * hi;
            ob[((size_t)(b * Sn + q)) * Dn + h * DHn + t * 16 + lm] =
                (__bf16)(acc[t][i] * inv[i]);
        }
}

// ---------------- residual + LayerNorm (fp32), emits fp32 + bf16 ----------------
__global__ __launch_bounds__(256) void k_rln(const float* __restrict__ xin,
                                             const float* __restrict__ t,
                                             const float* __restrict__ w,
                                             const float* __restrict__ bia,
                                             float* __restrict__ dst,
                                             __bf16* __restrict__ xb) {
    const int row = blockIdx.x, tid = threadIdx.x;
    __shared__ float red[256];
    const size_t base = (size_t)row * Dn;
    float v0 = xin[base + tid]       + t[base + tid];
    float v1 = xin[base + tid + 256] + t[base + tid + 256];
    red[tid] = v0 + v1;
    __syncthreads();
    for (int o = 128; o > 0; o >>= 1) { if (tid < o) red[tid] += red[tid + o]; __syncthreads(); }
    const float mean = red[0] * (1.0f / Dn);
    __syncthreads();
    const float d0 = v0 - mean, d1 = v1 - mean;
    red[tid] = d0 * d0 + d1 * d1;
    __syncthreads();
    for (int o = 128; o > 0; o >>= 1) { if (tid < o) red[tid] += red[tid + o]; __syncthreads(); }
    const float rs = rsqrtf(red[0] * (1.0f / Dn) + EPSc);
    const float y0 = d0 * rs * w[tid]       + bia[tid];
    const float y1 = d1 * rs * w[tid + 256] + bia[tid + 256];
    dst[base + tid] = y0;       dst[base + tid + 256] = y1;
    xb[base + tid] = (__bf16)y0; xb[base + tid + 256] = (__bf16)y1;
}

// ---------------- host orchestration ----------------
extern "C" void kernel_launch(void* const* d_in, const int* in_sizes, int n_in,
                              void* d_out, int out_size, void* d_ws, size_t ws_size,
                              hipStream_t stream) {
    (void)in_sizes; (void)n_in; (void)out_size; (void)ws_size;
    const float* img    = (const float*)d_in[0];
    const float* centre = (const float*)d_in[2];
    const float* Wq  = (const float*)d_in[3];
    const float* Wk  = (const float*)d_in[4];
    const float* Wv  = (const float*)d_in[5];
    const float* bq  = (const float*)d_in[6];
    const float* bk  = (const float*)d_in[7];
    const float* bv  = (const float*)d_in[8];
    const float* Wo  = (const float*)d_in[9];
    const float* bo  = (const float*)d_in[10];
    const float* W1  = (const float*)d_in[11];
    const float* b1  = (const float*)d_in[12];
    const float* W2  = (const float*)d_in[13];
    const float* b2  = (const float*)d_in[14];
    const float* ln1w = (const float*)d_in[15];
    const float* ln1b = (const float*)d_in[16];
    const float* ln2w = (const float*)d_in[17];
    const float* ln2b = (const float*)d_in[18];

    char* ws = (char*)d_ws;
    size_t off = 0;
    auto carve = [&](size_t bytes) -> char* {
        char* p = ws + off;
        off += (bytes + 255) & ~(size_t)255;
        return p;
    };
    const size_t DD = (size_t)Dn * Dn, DF = (size_t)Dn * FFn;
    __bf16* wqT = (__bf16*)carve(Ln * DD * 2);
    __bf16* wkT = (__bf16*)carve(Ln * DD * 2);
    __bf16* wvT = (__bf16*)carve(Ln * DD * 2);
    __bf16* woT = (__bf16*)carve(Ln * DD * 2);
    __bf16* w1T = (__bf16*)carve(Ln * DF * 2);   // [D][FF]
    __bf16* w2T = (__bf16*)carve(Ln * DF * 2);   // [FF][D]
    float*  xf  = (float*) carve((size_t)Ntok * Dn * 4);
    float*  tmp = (float*) carve((size_t)Ntok * Dn * 4);
    __bf16* xb  = (__bf16*)carve((size_t)Ntok * Dn * 2);
    __bf16* qbf = (__bf16*)carve((size_t)Ntok * Dn * 2);
    __bf16* ktb = (__bf16*)carve((size_t)Ntok * Dn * 2);   // [B][D][S]
    __bf16* vbf = (__bf16*)carve((size_t)Ntok * Dn * 2);
    __bf16* abf = (__bf16*)carve((size_t)Ntok * Dn * 2);
    __bf16* hbf = (__bf16*)carve((size_t)Ntok * FFn * 2);
    float*  scb = (float*) carve(Kc * 4);
    int*    lab = (int*)   carve(Ntok * 4);

    // mask labels
    k_centre_sums<<<1, Kc, 0, stream>>>(centre, scb);
    k_labels<<<Ntok / 256, 256, 0, stream>>>(img, scb, lab);
    // fp32 residual stream + bf16 activations
    k_init_x<<<(Ntok * Dn) / 256, 256, 0, stream>>>(img, xf, xb, Ntok * Dn);

    // one-time weight convert+transpose (bf16, B-operand friendly)
    for (int i = 0; i < Ln; ++i) {
        k_wt<<<(int)(DD / 256), 256, 0, stream>>>(Wq + i * DD, wqT + i * DD, Dn, Dn);
        k_wt<<<(int)(DD / 256), 256, 0, stream>>>(Wk + i * DD, wkT + i * DD, Dn, Dn);
        k_wt<<<(int)(DD / 256), 256, 0, stream>>>(Wv + i * DD, wvT + i * DD, Dn, Dn);
        k_wt<<<(int)(DD / 256), 256, 0, stream>>>(Wo + i * DD, woT + i * DD, Dn, Dn);
        k_wt<<<(int)(DF / 256), 256, 0, stream>>>(W1 + i * DF, w1T + i * DF, FFn, Dn);
        k_wt<<<(int)(DF / 256), 256, 0, stream>>>(W2 + i * DF, w2T + i * DF, Dn, FFn);
    }

    const int gD  = ((Ntok / 32) * (Dn / 64) + 7) / 8;   // 32x64 tiles, 8 waves/block
    const int gFF = ((Ntok / 32) * (FFn / 64) + 7) / 8;

    for (int i = 0; i < Ln; ++i) {
        const __bf16 *wq = wqT + i * DD, *wk = wkT + i * DD, *wv = wvT + i * DD,
                     *wo = woT + i * DD, *w1 = w1T + i * DF, *w2 = w2T + i * DF;
        // QKV projections (K goes out transposed)
        k_gemm<ST_BF16, false><<<gD, 256, 0, stream>>>(xb, wq, bq + i * Dn, qbf, Ntok, Dn, Dn);
        k_gemm<ST_KT,   false><<<gD, 256, 0, stream>>>(xb, wk, bk + i * Dn, ktb, Ntok, Dn, Dn);
        k_gemm<ST_BF16, false><<<gD, 256, 0, stream>>>(xb, wv, bv + i * Dn, vbf, Ntok, Dn, Dn);
        // masked flash attention
        k_attn<<<Bn * Hn * (Sn / 16), 32, 0, stream>>>(qbf, ktb, vbf, lab, abf);
        // output projection + residual LN1
        k_gemm<ST_F32,  false><<<gD, 256, 0, stream>>>(abf, wo, bo + i * Dn, tmp, Ntok, Dn, Dn);
        k_rln<<<Ntok, 256, 0, stream>>>(xf, tmp, ln1w + i * Dn, ln1b + i * Dn, xf, xb);
        // FFN
        k_gemm<ST_BF16, true ><<<gFF, 256, 0, stream>>>(xb, w1, b1 + i * FFn, hbf, Ntok, FFn, Dn);
        k_gemm<ST_F32,  false><<<gD, 256, 0, stream>>>(hbf, w2, b2 + i * Dn, tmp, Ntok, Dn, FFn);
        float* dst = (i == Ln - 1) ? (float*)d_out : xf;
        k_rln<<<Ntok, 256, 0, stream>>>(xf, tmp, ln2w + i * Dn, ln2b + i * Dn, dst, xb);
    }
}
